// MultiHeadAttention_14800457302282
// MI455X (gfx1250) — matrix operand order repaired and checked by reading, not compiled, not run on hardware
//
#include <hip/hip_runtime.h>

typedef __attribute__((ext_vector_type(16))) __bf16 v16bf;
typedef __attribute__((ext_vector_type(8)))  float  v8f;

union Frag16 { v16bf v; uint4 u[2]; };
union Pack8  { uint4 u; __bf16 h[8]; };

#define D_MODEL 1024
#define SEQ     2048
#define HEADS   16
#define HDIM    64

// ---------------- fragment loaders (per ISA 7.12.2 layouts) ----------------

// A-matrix 16x32 bf16, row-major source, row stride ld (elements).
__device__ inline v16bf load_a_frag(const __bf16* __restrict__ base, int ld) {
  int lane = threadIdx.x & 31;
  int half = lane >> 4, r = lane & 15;
  const __bf16* p = base + r * ld + half * 8;
  Frag16 f;
  f.u[0] = *(const uint4*)(p);
  f.u[1] = *(const uint4*)(p + 16);
  return f.v;
}

// B-matrix 32x16 bf16 where element(kk, n) = base[n*ld + kk] (kk contiguous).
__device__ inline v16bf load_b_frag(const __bf16* __restrict__ base, int ld) {
  int lane = threadIdx.x & 31;
  int half = lane >> 4, n = lane & 15;
  const __bf16* p = base + n * ld + half * 8;
  Frag16 f;
  f.u[0] = *(const uint4*)(p);
  f.u[1] = *(const uint4*)(p + 16);
  return f.v;
}

__device__ inline v8f wmma_bf16(v16bf a, v16bf b, v8f c) {
  return __builtin_amdgcn_wmma_f32_16x16x32_bf16(false, a, false, b, (short)0, c,
                                                 false, false);
}

// ---------------- conversion kernels ----------------

__global__ void cvt_f32_bf16(const float* __restrict__ in, __bf16* __restrict__ out, int n) {
  int i = blockIdx.x * 256 + threadIdx.x;
  if (i < n) out[i] = (__bf16)in[i];
}

// W [K,N] f32 row-major -> Wt [N,K] bf16 row-major
__global__ void cvt_transpose(const float* __restrict__ w, __bf16* __restrict__ wT,
                              int K, int N) {
  int i = blockIdx.x * 256 + threadIdx.x;
  if (i < K * N) {
    int k = i / N, n = i % N;
    wT[(size_t)n * K + k] = (__bf16)w[i];
  }
}

// ---------------- generic pipelined 32Mx64N wave-tile GEMM body ----------------
// Computes acc0/acc1 (2 x 4 16x16 tiles) for rows [mbase,mbase+32), cols [nbase,nbase+64).
// Ping-pong double buffering: buffers written directly by loads, no rotation copies.
template <typename EPI>
__device__ inline void gemm_body(const __bf16* __restrict__ A,
                                 const __bf16* __restrict__ wT,
                                 int mbase, int nbase, EPI epilogue) {
  const __bf16* Arow0 = A + (size_t)mbase * D_MODEL;
  const __bf16* Arow1 = A + (size_t)(mbase + 16) * D_MODEL;
  const __bf16* B0 = wT + (size_t)(nbase + 0)  * D_MODEL;
  const __bf16* B1 = wT + (size_t)(nbase + 16) * D_MODEL;
  const __bf16* B2 = wT + (size_t)(nbase + 32) * D_MODEL;
  const __bf16* B3 = wT + (size_t)(nbase + 48) * D_MODEL;

  v8f acc0[4], acc1[4];
#pragma unroll
  for (int t = 0; t < 4; ++t) { acc0[t] = {}; acc1[t] = {}; }

  // stage-A buffers (k = 0)
  v16bf a0A = load_a_frag(Arow0, D_MODEL);
  v16bf a1A = load_a_frag(Arow1, D_MODEL);
  v16bf bA0 = load_b_frag(B0, D_MODEL);
  v16bf bA1 = load_b_frag(B1, D_MODEL);
  v16bf bA2 = load_b_frag(B2, D_MODEL);
  v16bf bA3 = load_b_frag(B3, D_MODEL);

#pragma unroll 1
  for (int k0 = 0; k0 < D_MODEL; k0 += 64) {
    if (k0 + 256 < D_MODEL) {
      __builtin_prefetch(Arow0 + k0 + 256, 0, 3);
      __builtin_prefetch(B0 + k0 + 256, 0, 3);
      __builtin_prefetch(B2 + k0 + 256, 0, 3);
    }
    // stage B loads (k0+32)
    int kb = k0 + 32;
    v16bf a0B = load_a_frag(Arow0 + kb, D_MODEL);
    v16bf a1B = load_a_frag(Arow1 + kb, D_MODEL);
    v16bf bB0 = load_b_frag(B0 + kb, D_MODEL);
    v16bf bB1 = load_b_frag(B1 + kb, D_MODEL);
    v16bf bB2 = load_b_frag(B2 + kb, D_MODEL);
    v16bf bB3 = load_b_frag(B3 + kb, D_MODEL);

    // compute with stage A
    acc0[0] = wmma_bf16(a0A, bA0, acc0[0]);
    acc0[1] = wmma_bf16(a0A, bA1, acc0[1]);
    acc0[2] = wmma_bf16(a0A, bA2, acc0[2]);
    acc0[3] = wmma_bf16(a0A, bA3, acc0[3]);
    acc1[0] = wmma_bf16(a1A, bA0, acc1[0]);
    acc1[1] = wmma_bf16(a1A, bA1, acc1[1]);
    acc1[2] = wmma_bf16(a1A, bA2, acc1[2]);
    acc1[3] = wmma_bf16(a1A, bA3, acc1[3]);

    // stage A loads for next iteration (k0+64; last iteration reloads k0 harmlessly)
    int kn = (k0 + 64 < D_MODEL) ? k0 + 64 : k0;
    a0A = load_a_frag(Arow0 + kn, D_MODEL);
    a1A = load_a_frag(Arow1 + kn, D_MODEL);
    bA0 = load_b_frag(B0 + kn, D_MODEL);
    bA1 = load_b_frag(B1 + kn, D_MODEL);
    bA2 = load_b_frag(B2 + kn, D_MODEL);
    bA3 = load_b_frag(B3 + kn, D_MODEL);

    // compute with stage B
    acc0[0] = wmma_bf16(a0B, bB0, acc0[0]);
    acc0[1] = wmma_bf16(a0B, bB1, acc0[1]);
    acc0[2] = wmma_bf16(a0B, bB2, acc0[2]);
    acc0[3] = wmma_bf16(a0B, bB3, acc0[3]);
    acc1[0] = wmma_bf16(a1B, bB0, acc1[0]);
    acc1[1] = wmma_bf16(a1B, bB1, acc1[1]);
    acc1[2] = wmma_bf16(a1B, bB2, acc1[2]);
    acc1[3] = wmma_bf16(a1B, bB3, acc1[3]);
  }
  epilogue(acc0, acc1);
}

// ---------------- QKV projection GEMM ----------------
__global__ void __launch_bounds__(256) qkv_gemm_kernel(
    const __bf16* __restrict__ xb, const __bf16* __restrict__ wT,
    const float* __restrict__ bias,
    __bf16* __restrict__ qb, __bf16* __restrict__ kb, __bf16* __restrict__ vb) {
  int wave = threadIdx.x >> 5;
  int lane = threadIdx.x & 31;
  int mbase = blockIdx.y * 256 + wave * 32;
  int nbase = blockIdx.x * 64;
  int half = lane >> 4, nl = lane & 15;

  gemm_body(xb, wT, mbase, nbase, [&](v8f* acc0, v8f* acc1) {
#pragma unroll
    for (int mi = 0; mi < 2; ++mi) {
#pragma unroll
      for (int t = 0; t < 4; ++t) {
#pragma unroll
        for (int r = 0; r < 8; ++r) {
          int M = mbase + mi * 16 + r + half * 8;  // global row: b*2048 + s
          int N = nbase + 16 * t + nl;             // global col in [0,3072)
          float val = (mi ? acc1[t][r] : acc0[t][r]) + bias[N];
          __bf16 bv = (__bf16)val;
          int bb = M >> 11, s = M & 2047;
          int which = N >> 10;
          int rem = N & 1023;
          int hh = rem >> 6, hd = rem & 63;
          size_t headbase = (size_t)(bb * HEADS + hh);
          if (which == 0)
            qb[(headbase * SEQ + s) * HDIM + hd] = bv;
          else if (which == 1)
            kb[(headbase * SEQ + s) * HDIM + hd] = bv;
          else
            vb[(headbase * HDIM + hd) * SEQ + s] = bv;  // V transposed [b,h,d,s]
        }
      }
    }
  });
}

// ---------------- output projection GEMM ----------------
__global__ void __launch_bounds__(256) out_gemm_kernel(
    const __bf16* __restrict__ ab, const __bf16* __restrict__ wT,
    const float* __restrict__ bias, float* __restrict__ out) {
  int wave = threadIdx.x >> 5;
  int lane = threadIdx.x & 31;
  int mbase = blockIdx.y * 256 + wave * 32;
  int nbase = blockIdx.x * 64;
  int half = lane >> 4, nl = lane & 15;

  gemm_body(ab, wT, mbase, nbase, [&](v8f* acc0, v8f* acc1) {
#pragma unroll
    for (int mi = 0; mi < 2; ++mi) {
#pragma unroll
      for (int t = 0; t < 4; ++t) {
#pragma unroll
        for (int r = 0; r < 8; ++r) {
          int M = mbase + mi * 16 + r + half * 8;
          int N = nbase + 16 * t + nl;
          out[(size_t)M * D_MODEL + N] = (mi ? acc1[t][r] : acc0[t][r]) + bias[N];
        }
      }
    }
  });
}

// ---------------- fused attention kernel ----------------
// One workgroup (4 waves) per (b,h, 16-row q tile). LDS holds 16x2048 bf16 score strip.
__global__ void __launch_bounds__(128) attn_kernel(
    const __bf16* __restrict__ qb, const __bf16* __restrict__ kb,
    const __bf16* __restrict__ vb, __bf16* __restrict__ ob) {
  extern __shared__ char smem[];
  __bf16* P    = (__bf16*)smem;                 // 16*2048*2 = 65536 B
  float*  red  = (float*)(smem + 65536);        // 512 B
  float*  rowM = (float*)(smem + 66048);        // 64 B
  float*  rowL = (float*)(smem + 66112);        // 64 B

  int tile = blockIdx.x;            // 0..4095
  int q0   = (tile & 127) * 16;     // q-row base
  int bh   = tile >> 7;             // 0..31
  const __bf16* Q  = qb + (size_t)bh * SEQ * HDIM;
  const __bf16* K  = kb + (size_t)bh * SEQ * HDIM;
  const __bf16* Vt = vb + (size_t)bh * HDIM * SEQ;

  int wave = threadIdx.x >> 5;
  int lane = threadIdx.x & 31;
  int half = lane >> 4, nl = lane & 15;

  // Q fragments (K-dim = 64 -> two 16x32 fragments)
  v16bf qf0 = load_a_frag(Q + (size_t)q0 * HDIM + 0,  HDIM);
  v16bf qf1 = load_a_frag(Q + (size_t)q0 * HDIM + 32, HDIM);

  // ---- phase 1: scores = (Q K^T)/8 ; wave w covers keys [w*512, w*512+512)
  {
    int base = wave * 512;
    v16bf b0A = load_b_frag(K + (size_t)base * HDIM + 0,  HDIM);
    v16bf b1A = load_b_frag(K + (size_t)base * HDIM + 32, HDIM);
#pragma unroll 1
    for (int kt = 0; kt < 32; kt += 2) {
      int kcA = base + kt * 16;
      int kcB = kcA + 16;
      // stage B loads
      v16bf b0B = load_b_frag(K + (size_t)kcB * HDIM + 0,  HDIM);
      v16bf b1B = load_b_frag(K + (size_t)kcB * HDIM + 32, HDIM);
      // compute + store stage A
      v8f acc = {};
      acc = wmma_bf16(qf0, b0A, acc);
      acc = wmma_bf16(qf1, b1A, acc);
#pragma unroll
      for (int r = 0; r < 8; ++r)
        P[(r + half * 8) * SEQ + kcA + nl] = (__bf16)(acc[r] * 0.125f);
      // stage A loads for next iteration
      int kn = (kt + 2 < 32) ? kcA + 32 : kcA;
      b0A = load_b_frag(K + (size_t)kn * HDIM + 0,  HDIM);
      b1A = load_b_frag(K + (size_t)kn * HDIM + 32, HDIM);
      // compute + store stage B
      v8f acc2 = {};
      acc2 = wmma_bf16(qf0, b0B, acc2);
      acc2 = wmma_bf16(qf1, b1B, acc2);
#pragma unroll
      for (int r = 0; r < 8; ++r)
        P[(r + half * 8) * SEQ + kcB + nl] = (__bf16)(acc2[r] * 0.125f);
    }
  }
  __syncthreads();

  // ---- phase 2: softmax per row (8 threads/row, 256 cols each, 16B vectorized)
  {
    int row = threadIdx.x >> 3;
    int seg = threadIdx.x & 7;
    __bf16* prow = P + row * SEQ + seg * 256;
    float lmax = -1e30f;
    for (int c = 0; c < 256; c += 8) {
      Pack8 pk; pk.u = *(const uint4*)(prow + c);
#pragma unroll
      for (int j = 0; j < 8; ++j) lmax = fmaxf(lmax, (float)pk.h[j]);
    }
    red[row * 8 + seg] = lmax;
    __syncthreads();
    if (seg == 0) {
      float m = red[row * 8];
      for (int j = 1; j < 8; ++j) m = fmaxf(m, red[row * 8 + j]);
      rowM[row] = m;
    }
    __syncthreads();
    float m = rowM[row];
    float lsum = 0.f;
    for (int c = 0; c < 256; c += 8) {
      Pack8 pk; pk.u = *(const uint4*)(prow + c);
#pragma unroll
      for (int j = 0; j < 8; ++j) {
        float p = __expf((float)pk.h[j] - m);
        pk.h[j] = (__bf16)p;
        lsum += p;
      }
      *(uint4*)(prow + c) = pk.u;
    }
    red[row * 8 + seg] = lsum;
    __syncthreads();
    if (seg == 0) {
      float l = 0.f;
      for (int j = 0; j < 8; ++j) l += red[row * 8 + j];
      rowL[row] = l;
    }
    __syncthreads();
  }

  // ---- phase 3: O = P @ V ; wave w owns hd columns [w*16, w*16+16), ping-pong
  int n0 = wave * 16;
  const __bf16* Vcol = Vt + (size_t)n0 * SEQ;
  v8f oacc = {};
  {
    v16bf aA = load_a_frag(&P[0], SEQ);
    v16bf bA = load_b_frag(Vcol, SEQ);
#pragma unroll 1
    for (int k0 = 0; k0 < SEQ; k0 += 64) {
      if (k0 + 256 < SEQ) __builtin_prefetch(Vcol + k0 + 256, 0, 3);
      // stage B
      v16bf aB = load_a_frag(&P[k0 + 32], SEQ);
      v16bf bB = load_b_frag(Vcol + k0 + 32, SEQ);
      oacc = wmma_bf16(aA, bA, oacc);
      // stage A for next iteration
      int kn = (k0 + 64 < SEQ) ? k0 + 64 : k0;
      aA = load_a_frag(&P[kn], SEQ);
      bA = load_b_frag(Vcol + kn, SEQ);
      oacc = wmma_bf16(aB, bB, oacc);
    }
  }

  int b_ = bh >> 4, h_ = bh & 15;
#pragma unroll
  for (int r = 0; r < 8; ++r) {
    int M = r + half * 8;
    float val = oacc[r] / rowL[M];
    size_t row = (size_t)(b_ * SEQ + q0 + M);
    ob[row * D_MODEL + h_ * HDIM + n0 + nl] = (__bf16)val;  // [B,S,H*Hd]
  }
}

// ---------------- launcher ----------------

extern "C" void kernel_launch(void* const* d_in, const int* in_sizes, int n_in,
                              void* d_out, int out_size, void* d_ws, size_t ws_size,
                              hipStream_t stream) {
  const float* x    = (const float*)d_in[0];  // [2,2048,1024]
  const float* Wqkv = (const float*)d_in[1];  // [1024,3072]
  const float* bqkv = (const float*)d_in[2];  // [3072]
  const float* Wout = (const float*)d_in[3];  // [1024,1024]
  const float* bout = (const float*)d_in[4];  // [1024]
  float* out = (float*)d_out;                 // [2,2048,1024]

  char* ws = (char*)d_ws;
  const size_t MB = 1024 * 1024;
  __bf16* xb    = (__bf16*)(ws + 0 * MB);   // 8 MB   [4096,1024]
  __bf16* wqkvT = (__bf16*)(ws + 8 * MB);   // 6 MB   [3072,1024]
  __bf16* woutT = (__bf16*)(ws + 14 * MB);  // 2 MB   [1024,1024]
  __bf16* qb    = (__bf16*)(ws + 16 * MB);  // 8 MB   [B,H,S,Hd]
  __bf16* kb    = (__bf16*)(ws + 24 * MB);  // 8 MB   [B,H,S,Hd]
  __bf16* vb    = (__bf16*)(ws + 32 * MB);  // 8 MB   [B,H,Hd,S] (transposed)
  __bf16* attnb = (__bf16*)(ws + 40 * MB);  // 8 MB   [B*S, 1024]

  const int NX = 2 * SEQ * D_MODEL;  // 4,194,304
  cvt_f32_bf16<<<(NX + 255) / 256, 256, 0, stream>>>(x, xb, NX);
  cvt_transpose<<<(D_MODEL * 3 * D_MODEL + 255) / 256, 256, 0, stream>>>(
      Wqkv, wqkvT, D_MODEL, 3 * D_MODEL);
  cvt_transpose<<<(D_MODEL * D_MODEL + 255) / 256, 256, 0, stream>>>(
      Wout, woutT, D_MODEL, D_MODEL);

  // M=4096 (256 rows/WG), N=3072: grid (N/64, M/256)
  qkv_gemm_kernel<<<dim3(48, 16), 256, 0, stream>>>(xb, wqkvT, bqkv, qb, kb, vb);

  // B*H*(S/16) = 4096 tiles; 66176 B dynamic LDS
  attn_kernel<<<4096, 128, 66176, stream>>>(qb, kb, vb, attnb);

  // M=4096 (256 rows/WG), N=1024: grid (N/64, M/256)
  out_gemm_kernel<<<dim3(16, 16), 256, 0, stream>>>(attnb, woutT, bout, out);
}
